// HyenaFilter_54520314855701
// MI455X (gfx1250) — matrix operand order repaired and checked by reading, not compiled, not run on hardware
//
#include <hip/hip_runtime.h>
#include <hip/hip_bf16.h>
#include <math.h>

#define D_MODEL 1024
#define SEQ_LEN 8192
#define ORDER   16
#define BATCH   4

typedef __attribute__((ext_vector_type(16))) __bf16 v16bf;
typedef __attribute__((ext_vector_type(8)))  float  v8f;

#if defined(__gfx1250__) && __has_builtin(__builtin_amdgcn_tensor_load_to_lds) && __has_builtin(__builtin_amdgcn_s_wait_tensorcnt)
#define USE_TDM 1
typedef unsigned int v4u __attribute__((ext_vector_type(4)));
typedef int          v8i __attribute__((ext_vector_type(8)));
typedef int          v4i __attribute__((ext_vector_type(4)));
#else
#define USE_TDM 0
#endif

__device__ __forceinline__ unsigned short f32_to_bf16(float f) {
  unsigned int u = __float_as_uint(f);
  u += 0x7FFFu + ((u >> 16) & 1u);   // round-to-nearest-even
  return (unsigned short)(u >> 16);
}

// ---------------------------------------------------------------------------
// Kernel 1: k[c][t] = decay(c,t) * MLP(z[t])  -> bf16, channel-major in d_ws
// ---------------------------------------------------------------------------
__global__ __launch_bounds__(256) void hyena_filter_build(
    const float* __restrict__ z,      // (SEQ_LEN, 3)
    const float* __restrict__ W1,     // (3, 16)
    const float* __restrict__ b1,     // (16)
    const float* __restrict__ W2,     // (16, 16)
    const float* __restrict__ b2,     // (16)
    const float* __restrict__ W3,     // (16, 16)
    const float* __restrict__ b3,     // (16)
    const float* __restrict__ Wout,   // (16, 1024)
    const float* __restrict__ freq,   // (16)
    unsigned short* __restrict__ kbf) // (1024, 8192) bf16
{
  const int t   = blockIdx.x;
  const int tid = threadIdx.x;

  const float z0 = z[t * 3 + 0];
  const float z1 = z[t * 3 + 1];
  const float z2 = z[t * 3 + 2];

  float h1[ORDER], h2[ORDER], h3[ORDER];
  #pragma unroll
  for (int j = 0; j < ORDER; ++j) {
    float a = z0 * W1[0 * ORDER + j] + z1 * W1[1 * ORDER + j] + z2 * W1[2 * ORDER + j] + b1[j];
    h1[j] = __sinf(freq[j] * a);
  }
  #pragma unroll
  for (int j = 0; j < ORDER; ++j) {
    float a = b2[j];
    #pragma unroll
    for (int i = 0; i < ORDER; ++i) a += h1[i] * W2[i * ORDER + j];
    h2[j] = __sinf(freq[j] * a);
  }
  #pragma unroll
  for (int j = 0; j < ORDER; ++j) {
    float a = b3[j];
    #pragma unroll
    for (int i = 0; i < ORDER; ++i) a += h2[i] * W3[i * ORDER + j];
    h3[j] = __sinf(freq[j] * a);
  }

  const float tlin      = (float)t / (float)(SEQ_LEN - 1);
  const float min_decay = -3.0701134573253944f;   // log(0.01)/1.5
  const float max_decay = -15.350567286626972f;   // log(0.01)/0.3

  #pragma unroll
  for (int cc = 0; cc < 4; ++cc) {
    const int c = tid * 4 + cc;
    const float delta = min_decay + (max_decay - min_decay) * (float)c / (float)(D_MODEL - 1);
    const float dec   = __expf(-tlin * fabsf(delta));
    float kv = 0.0f;
    #pragma unroll
    for (int i = 0; i < ORDER; ++i) kv += h3[i] * Wout[i * D_MODEL + c];
    kbf[(size_t)c * SEQ_LEN + t] = f32_to_bf16(kv * dec);
  }
}

// ---------------------------------------------------------------------------
// Kernel 2: causal Toeplitz convolution as WMMA bf16 GEMM + bias skip.
// Block = 256 threads (8 wave32) <-> one (batch, channel, quarter).
// Wave g owns output positions [256g, 256g+256): 16 output blocks as N-cols.
// Lags processed in chunks of 8 lag-pairs:
//   stage: 256 threads build the chunk's 8 Toeplitz A-fragments into LDS
//          (fragment-major, so compute reads them with aligned ds_load_b128)
//   compute: per wave, 8x { A: 2xb128, B: 2xb128 slab of x, v_wmma bf16 }
// ---------------------------------------------------------------------------
__global__ __launch_bounds__(256) void hyena_conv_wmma(
    const float* __restrict__ x,            // (B, C, L) f32
    const unsigned short* __restrict__ kbf, // (C, L)    bf16
    const float* __restrict__ bias,         // (C)
    float* __restrict__ out)                // (B, C, L) f32
{
  // k padded: [0..15] = zeros, [16..16+8191] = k_c, tail zeros.
  __shared__ __align__(32) unsigned short k_pad[16 + SEQ_LEN + 32];
  __shared__ __align__(32) unsigned short x_lds[SEQ_LEN];
  __shared__ __align__(32) unsigned short a_lds[8 * 32 * 16];   // 8 A-fragments

  const int blk = blockIdx.x;
  const int qb  = blk & 3;                  // quarter of channel
  const int c   = (blk >> 2) & (D_MODEL - 1);
  const int b   = blk >> 12;
  const int tid = threadIdx.x;

  const size_t chbase = ((size_t)b * D_MODEL + c) * SEQ_LEN;
  const unsigned short* kc = kbf + (size_t)c * SEQ_LEN;

  // ---- stage k_c into LDS (TDM if available), x_c as f32->bf16 ----
#if USE_TDM
  if (tid == 0) {
    const unsigned long long ga  = (unsigned long long)kc;
    const unsigned           lds = (unsigned)(unsigned long long)(void*)&k_pad[16];
    v4u g0;
    g0.x = 1u;                                    // count=1, user descriptor
    g0.y = lds;                                   // lds_addr (bytes)
    g0.z = (unsigned)(ga & 0xFFFFFFFFu);          // global_addr[31:0]
    g0.w = (unsigned)(ga >> 32) | 0x80000000u;    // global_addr[56:32] | type=2
    v8i g1;
    g1[0] = 0x00010000;                           // data_size=1 (2 bytes)
    g1[1] = (int)(SEQ_LEN << 16);                 // tensor_dim0 = 8192 (low16)
    g1[2] = (int)(1 << 16);                       // tensor_dim1 = 1
    g1[3] = (int)(SEQ_LEN << 16);                 // tile_dim0 = 8192
    g1[4] = 1;                                    // tile_dim1 = 1
    g1[5] = SEQ_LEN;                              // tensor_dim0_stride = 8192
    g1[6] = 0;
    g1[7] = 0;
    v4i g2 = {0, 0, 0, 0};
    v4i g3 = {0, 0, 0, 0};
    v8i g4 = {0, 0, 0, 0, 0, 0, 0, 0};            // 6-arg toolchain extra group
    __builtin_amdgcn_tensor_load_to_lds(g0, g1, g2, g3, g4, 0);
    __builtin_amdgcn_s_wait_tensorcnt(0);
  }
  if (tid < 16)       k_pad[tid] = 0;
  else if (tid < 48)  k_pad[16 + SEQ_LEN + (tid - 16)] = 0;
#else
  for (int i = tid; i < SEQ_LEN; i += 256) k_pad[16 + i] = kc[i];
  if (tid < 16)       k_pad[tid] = 0;
  else if (tid < 48)  k_pad[16 + SEQ_LEN + (tid - 16)] = 0;
#endif
  for (int i = tid; i < SEQ_LEN; i += 256) {
    __builtin_prefetch(&x[chbase + i + 2048], 0, 0);
    x_lds[i] = f32_to_bf16(x[chbase + i]);
  }
  __syncthreads();

  const int wave = tid >> 5;
  const int lane = tid & 31;
  const int n    = lane & 15;     // output-block column (N) / A row (M)
  const int half = lane >> 4;     // K-half of the fragment
  const int g    = qb * 8 + wave; // wave's output group (0..31)

  union AB { v16bf v; uint4 q[2]; unsigned short u[16]; };

  v8f acc = {};
  const int ndpair = 8 * (g + 1);     // this wave's causal lag-pair count
  const int nch    = 8 * qb + 8;      // chunks needed by the block's max wave

  for (int ch = 0; ch < nch; ++ch) {
    // ---- cooperative staging of 8 Toeplitz A-fragments ----
    {
      const int j  = tid >> 5;        // tile in chunk
      const int l  = tid & 31;
      const int sm = l & 15;
      const int sh = l >> 4;
      const int d0 = 2 * (ch * 8 + j);
      union AB t;
      #pragma unroll
      for (int e = 0; e < 16; ++e) {
        const int kk   = 16 * sh + e;
        const int kidx = (kk < 16) ? (16 * d0 + sm - kk) : (16 * d0 + 32 + sm - kk);
        t.u[e] = k_pad[16 + kidx];    // pads make all indices in-range
      }
      uint4* dst = (uint4*)&a_lds[tid * 16];
      dst[0] = t.q[0];
      dst[1] = t.q[1];
    }
    __syncthreads();

    // ---- compute: 8 WMMAs against the staged fragments ----
    if (ch * 8 < ndpair) {
      #pragma unroll
      for (int j = 0; j < 8; ++j) {
        const int d0 = 2 * (ch * 8 + j);
        union AB Am, Bm;
        const uint4* ap = (const uint4*)&a_lds[(j * 32 + lane) * 16];
        Am.q[0] = ap[0];
        Am.q[1] = ap[1];
        const int xb = 16 * (16 * g + n - d0 - half);
        if (xb >= 0) {
          const uint4* bp = (const uint4*)&x_lds[xb];   // 32B aligned
          Bm.q[0] = bp[0];
          Bm.q[1] = bp[1];
        } else {                                        // acausal -> zero
          Bm.q[0] = make_uint4(0, 0, 0, 0);
          Bm.q[1] = make_uint4(0, 0, 0, 0);
        }
        acc = __builtin_amdgcn_wmma_f32_16x16x32_bf16(
            false, Am.v, false, Bm.v, (short)0, acc, false, false);
      }
    }
    __syncthreads();
  }

  // ---- epilogue: y = conv + x * bias[c] ----
  const float bc = bias[c];
  #pragma unroll
  for (int r = 0; r < 8; ++r) {
    const int mm = r + 8 * half;
    const int p  = 256 * g + 16 * n + mm;
    out[chbase + p] = acc[r] + x[chbase + p] * bc;
  }
}

// ---------------------------------------------------------------------------
extern "C" void kernel_launch(void* const* d_in, const int* in_sizes, int n_in,
                              void* d_out, int out_size, void* d_ws, size_t ws_size,
                              hipStream_t stream) {
  const float* x    = (const float*)d_in[0];
  // d_in[1] = L (python int) -> fixed at SEQ_LEN here
  const float* z    = (const float*)d_in[2];
  const float* W1   = (const float*)d_in[3];
  const float* b1   = (const float*)d_in[4];
  const float* W2   = (const float*)d_in[5];
  const float* b2   = (const float*)d_in[6];
  const float* W3   = (const float*)d_in[7];
  const float* b3   = (const float*)d_in[8];
  const float* Wout = (const float*)d_in[9];
  const float* freq = (const float*)d_in[10];
  const float* bias = (const float*)d_in[11];
  float* out = (float*)d_out;

  unsigned short* kbf = (unsigned short*)d_ws;   // 1024*8192 bf16 = 16 MB

  hyena_filter_build<<<dim3(SEQ_LEN), dim3(256), 0, stream>>>(
      z, W1, b1, W2, b2, W3, b3, Wout, freq, kbf);

  hyena_conv_wmma<<<dim3(BATCH * D_MODEL * 4), dim3(256), 0, stream>>>(
      x, kbf, bias, out);
}
